// TransformerEncoder_18305150616299
// MI455X (gfx1250) — matrix operand order repaired and checked by reading, not compiled, not run on hardware
//
#include <hip/hip_runtime.h>
#include <hip/hip_bf16.h>

// ---------------------------------------------------------------------------
// Transformer-XL style encoder on gfx1250 (MI455X).
// All GEMMs run through v_wmma_f32_16x16x32_f16 (f16 inputs, f32 accumulate).
// ---------------------------------------------------------------------------

typedef __attribute__((ext_vector_type(16))) _Float16 v16h;
typedef __attribute__((ext_vector_type(8)))  _Float16 v8h;
typedef __attribute__((ext_vector_type(8)))  float    v8f;

#define NUM_LAYERS 4
#define D_MODEL    1024
#define N_HEAD     8
#define HEAD_DIM   128
#define BB         8
#define LL         512
#define BL         (BB * LL)          // 4096 rows
#define L2X        (2 * LL)           // 1024 relative positions

// ---------------------------------------------------------------------------
// WMMA fragment loaders (contiguous k order, same for A and B, wave32)
// A: 16x32 f16, row-major source.  lane m = l&15, half h = l>>4.
//    element e: K = k0 + 16*h + e  -> two contiguous v8h loads.
// ---------------------------------------------------------------------------
__device__ __forceinline__ v16h load_a16x32(const _Float16* __restrict__ A,
                                            long long lda, int row0, int k0,
                                            int lane) {
  const int m = lane & 15, hs = lane >> 4;
  const _Float16* p = A + (long long)(row0 + m) * lda + k0 + 16 * hs;
  v8h lo = *(const v8h*)p;          // K = k0+16h   .. +7
  v8h hi = *(const v8h*)(p + 8);    // K = k0+16h+8 .. +7
  return __builtin_shufflevector(lo, hi, 0, 1, 2, 3, 4, 5, 6, 7,
                                 8, 9, 10, 11, 12, 13, 14, 15);
}

// B for NT GEMM: logical B[k][n] = W[n][k], W row-major [N,K].
// lane n = l&15, h = l>>4; element e: K = k0 + 16*h + e -> one v16h load.
__device__ __forceinline__ v16h load_bt16x32(const _Float16* __restrict__ W,
                                             long long ldw, int col0, int k0,
                                             int lane) {
  const int n = lane & 15, hs = lane >> 4;
  return *(const v16h*)(W + (long long)(col0 + n) * ldw + k0 + 16 * hs);
}

// ---------------------------------------------------------------------------
// Batched NT GEMM: C[M,N] = A[M,K] * W[N,K]^T (+bias)(relu)
// Batch z decomposes as (b = z/nh, n = z%nh); per-operand (b,n) strides.
// Block: 256 threads = 8 waves; block tile 256(M) x 64(N); wave tile 32x64.
// K loop is software-pipelined (double-buffered fragments, peeled epilogue)
// so the 8 WMMAs of step k overlap the 6 fragment loads of step k+1.
// Requires: M % 32 == 0, N % 64 == 0, K % 32 == 0 (true for all launches).
// ---------------------------------------------------------------------------
__global__ __launch_bounds__(256) void gemm_nt_wmma(
    const _Float16* __restrict__ A, long long ldA, long long sbA, long long snA,
    const _Float16* __restrict__ W, long long ldW, long long sbW, long long snW,
    float* __restrict__ C, long long ldC, long long sbC, long long snC,
    int M, int N, int K, int nh,
    const float* __restrict__ bias, int relu) {
  const int lane = threadIdx.x & 31;
  const int wave = threadIdx.x >> 5;
  const int z = blockIdx.z;
  const int zb = z / nh, zn = z % nh;
  A += (long long)zb * sbA + (long long)zn * snA;
  W += (long long)zb * sbW + (long long)zn * snW;
  C += (long long)zb * sbC + (long long)zn * snC;

  const int row0 = blockIdx.y * 256 + wave * 32;   // two 16-row sub-tiles
  const int colBase = blockIdx.x * 64;
  if (row0 + 32 > M) return;                       // wave-uniform

  v8f acc0[4] = {};
  v8f acc1[4] = {};

  // prologue: fragments for k0 = 0
  v16h a0 = load_a16x32(A, ldA, row0,      0, lane);
  v16h a1 = load_a16x32(A, ldA, row0 + 16, 0, lane);
  v16h b0 = load_bt16x32(W, ldW, colBase,      0, lane);
  v16h b1 = load_bt16x32(W, ldW, colBase + 16, 0, lane);
  v16h b2 = load_bt16x32(W, ldW, colBase + 32, 0, lane);
  v16h b3 = load_bt16x32(W, ldW, colBase + 48, 0, lane);

#pragma unroll 2
  for (int k0 = 0; k0 + 32 < K; k0 += 32) {
    const int kn = k0 + 32;
    // issue next panel's loads before consuming the current fragments
    const v16h na0 = load_a16x32(A, ldA, row0,      kn, lane);
    const v16h na1 = load_a16x32(A, ldA, row0 + 16, kn, lane);
    const v16h nb0 = load_bt16x32(W, ldW, colBase,      kn, lane);
    const v16h nb1 = load_bt16x32(W, ldW, colBase + 16, kn, lane);
    const v16h nb2 = load_bt16x32(W, ldW, colBase + 32, kn, lane);
    const v16h nb3 = load_bt16x32(W, ldW, colBase + 48, kn, lane);
    if (kn + 32 < K) {
      // stream the panel after next through the cache (global_prefetch_b8)
      __builtin_prefetch((const void*)(A + (long long)(row0 + (lane & 15)) * ldA
                                       + kn + 32), 0, 3);
    }

    acc0[0] = __builtin_amdgcn_wmma_f32_16x16x32_f16(false, a0, false, b0,
                                                     (short)0, acc0[0], false, false);
    acc1[0] = __builtin_amdgcn_wmma_f32_16x16x32_f16(false, a1, false, b0,
                                                     (short)0, acc1[0], false, false);
    acc0[1] = __builtin_amdgcn_wmma_f32_16x16x32_f16(false, a0, false, b1,
                                                     (short)0, acc0[1], false, false);
    acc1[1] = __builtin_amdgcn_wmma_f32_16x16x32_f16(false, a1, false, b1,
                                                     (short)0, acc1[1], false, false);
    acc0[2] = __builtin_amdgcn_wmma_f32_16x16x32_f16(false, a0, false, b2,
                                                     (short)0, acc0[2], false, false);
    acc1[2] = __builtin_amdgcn_wmma_f32_16x16x32_f16(false, a1, false, b2,
                                                     (short)0, acc1[2], false, false);
    acc0[3] = __builtin_amdgcn_wmma_f32_16x16x32_f16(false, a0, false, b3,
                                                     (short)0, acc0[3], false, false);
    acc1[3] = __builtin_amdgcn_wmma_f32_16x16x32_f16(false, a1, false, b3,
                                                     (short)0, acc1[3], false, false);

    a0 = na0; a1 = na1; b0 = nb0; b1 = nb1; b2 = nb2; b3 = nb3;
  }

  // epilogue: last K-step
  acc0[0] = __builtin_amdgcn_wmma_f32_16x16x32_f16(false, a0, false, b0,
                                                   (short)0, acc0[0], false, false);
  acc1[0] = __builtin_amdgcn_wmma_f32_16x16x32_f16(false, a1, false, b0,
                                                   (short)0, acc1[0], false, false);
  acc0[1] = __builtin_amdgcn_wmma_f32_16x16x32_f16(false, a0, false, b1,
                                                   (short)0, acc0[1], false, false);
  acc1[1] = __builtin_amdgcn_wmma_f32_16x16x32_f16(false, a1, false, b1,
                                                   (short)0, acc1[1], false, false);
  acc0[2] = __builtin_amdgcn_wmma_f32_16x16x32_f16(false, a0, false, b2,
                                                   (short)0, acc0[2], false, false);
  acc1[2] = __builtin_amdgcn_wmma_f32_16x16x32_f16(false, a1, false, b2,
                                                   (short)0, acc1[2], false, false);
  acc0[3] = __builtin_amdgcn_wmma_f32_16x16x32_f16(false, a0, false, b3,
                                                   (short)0, acc0[3], false, false);
  acc1[3] = __builtin_amdgcn_wmma_f32_16x16x32_f16(false, a1, false, b3,
                                                   (short)0, acc1[3], false, false);

  const int cn = lane & 15, hs = lane >> 4;
#pragma unroll
  for (int t = 0; t < 4; ++t) {
    const int col = colBase + 16 * t + cn;
    const float badd = bias ? bias[col] : 0.f;
#pragma unroll
    for (int v = 0; v < 8; ++v) {
      const int r = row0 + v + 8 * hs;
      float val0 = acc0[t][v] + badd;
      float val1 = acc1[t][v] + badd;
      if (relu) {
        val0 = val0 > 0.f ? val0 : 0.f;
        val1 = val1 > 0.f ? val1 : 0.f;
      }
      C[(long long)r * ldC + col] = val0;
      C[(long long)(r + 16) * ldC + col] = val1;
    }
  }
}

// ---------------------------------------------------------------------------
// Relative-position term: B_[i,c] = q[i,:]·pe[c,:]  (WMMA NT, K=128),
// then the Transformer-XL shift folded as a scatter-add:
//   scores[i, c-L+i] += B_[i,c] + D_[head, c]   (unique (i,c)->(i,k) mapping)
// K=128 is fully unrolled; the scheduler can hoist all fragment loads.
// ---------------------------------------------------------------------------
__global__ __launch_bounds__(256) void relb_shift_add(
    const _Float16* __restrict__ Q,   // [B*H, L, 128]
    const _Float16* __restrict__ PE,  // [2L, 128]
    const float* __restrict__ Dn,     // [H, 2L]
    float* __restrict__ S,            // [B*H, L, L]
    int L, int H) {
  const int lane = threadIdx.x & 31;
  const int wave = threadIdx.x >> 5;
  const int z = blockIdx.z;
  const int nn = z % H;
  const _Float16* q = Q + (long long)z * L * HEAD_DIM;
  float* s = S + (long long)z * L * L;

  const int row0 = blockIdx.y * 256 + wave * 32;
  const int colBase = blockIdx.x * 64;
  if (row0 + 32 > L) return;

  v8f acc0[4] = {};
  v8f acc1[4] = {};
#pragma unroll
  for (int k0 = 0; k0 < HEAD_DIM; k0 += 32) {
    const v16h a0 = load_a16x32(q, HEAD_DIM, row0,      k0, lane);
    const v16h a1 = load_a16x32(q, HEAD_DIM, row0 + 16, k0, lane);
#pragma unroll
    for (int t = 0; t < 4; ++t) {
      const v16h b = load_bt16x32(PE, HEAD_DIM, colBase + 16 * t, k0, lane);
      acc0[t] = __builtin_amdgcn_wmma_f32_16x16x32_f16(
          false, a0, false, b, (short)0, acc0[t], false, false);
      acc1[t] = __builtin_amdgcn_wmma_f32_16x16x32_f16(
          false, a1, false, b, (short)0, acc1[t], false, false);
    }
  }

  const int cn = lane & 15, hs = lane >> 4;
#pragma unroll
  for (int t = 0; t < 4; ++t) {
    const int c = colBase + 16 * t + cn;                 // 0 .. 2L-1
    const float dadd = Dn[(long long)nn * 2 * L + c];
#pragma unroll
    for (int v = 0; v < 8; ++v) {
      const int i0 = row0 + v + 8 * hs;
      const int k0i = c - L + i0;                        // shifted key index
      if (k0i >= 0 && k0i < L)
        s[(long long)i0 * L + k0i] += acc0[t][v] + dadd;
      const int i1 = i0 + 16;
      const int k1i = c - L + i1;
      if (k1i >= 0 && k1i < L)
        s[(long long)i1 * L + k1i] += acc1[t][v] + dadd;
    }
  }
}

// ---------------------------------------------------------------------------
// Row softmax over scores (L=512 per row), mask applied, f16 probs out.
// ---------------------------------------------------------------------------
__global__ __launch_bounds__(256) void softmax_row(
    float* __restrict__ S, const int* __restrict__ mask,
    _Float16* __restrict__ P, int L, int H) {
  const int row = blockIdx.x;                 // (b*H + n)*L + i
  const int b = row / (H * L);
  float* s = S + (long long)row * L;
  _Float16* p = P + (long long)row * L;
  const int* mrow = mask + (long long)b * L;

  __shared__ float red[256];
  const int t = threadIdx.x;
  const int k0 = t, k1 = t + 256;             // L == 512, block == 256

  float v0 = (mrow[k0] == 0) ? -INFINITY : s[k0];
  float v1 = (mrow[k1] == 0) ? -INFINITY : s[k1];

  red[t] = fmaxf(v0, v1);
  __syncthreads();
  for (int o = 128; o > 0; o >>= 1) {
    if (t < o) red[t] = fmaxf(red[t], red[t + o]);
    __syncthreads();
  }
  const float bm = red[0];
  __syncthreads();

  const float e0 = __expf(v0 - bm);
  const float e1 = __expf(v1 - bm);
  red[t] = e0 + e1;
  __syncthreads();
  for (int o = 128; o > 0; o >>= 1) {
    if (t < o) red[t] += red[t + o];
    __syncthreads();
  }
  const float inv = 1.f / red[0];
  p[k0] = (_Float16)(e0 * inv);
  p[k1] = (_Float16)(e1 * inv);
}

// ---------------------------------------------------------------------------
// Fused residual add + LayerNorm; writes f32 (next residual) and f16 (GEMM A)
// ---------------------------------------------------------------------------
__global__ __launch_bounds__(256) void add_layernorm(
    const float* __restrict__ X, const float* __restrict__ Y,
    const float* __restrict__ g, const float* __restrict__ be,
    float* __restrict__ O32, _Float16* __restrict__ O16, int D) {
  const long long row = blockIdx.x;
  const float* x = X + row * D;
  const float* y = Y + row * D;
  __shared__ float red[256];
  const int t = threadIdx.x;

  float vals[4];
  float lsum = 0.f;
#pragma unroll
  for (int j = 0; j < 4; ++j) {
    const int d = t + j * 256;
    const float v = x[d] + y[d];
    vals[j] = v; lsum += v;
  }
  red[t] = lsum; __syncthreads();
  for (int o = 128; o > 0; o >>= 1) { if (t < o) red[t] += red[t + o]; __syncthreads(); }
  const float mu = red[0] / (float)D;
  __syncthreads();

  float lvar = 0.f;
#pragma unroll
  for (int j = 0; j < 4; ++j) { const float d0 = vals[j] - mu; lvar += d0 * d0; }
  red[t] = lvar; __syncthreads();
  for (int o = 128; o > 0; o >>= 1) { if (t < o) red[t] += red[t + o]; __syncthreads(); }
  const float rs = rsqrtf(red[0] / (float)D + 1e-5f);

#pragma unroll
  for (int j = 0; j < 4; ++j) {
    const int d = t + j * 256;
    const float o = (vals[j] - mu) * rs * g[d] + be[d];
    O32[row * D + d] = o;
    O16[row * D + d] = (_Float16)o;
  }
}

// ---------------------------------------------------------------------------
// Elementwise / prep kernels
// ---------------------------------------------------------------------------
__global__ void init_h(const float* __restrict__ x, _Float16* __restrict__ h16,
                       float* __restrict__ hcur, long long n) {
  const long long i = (long long)blockIdx.x * blockDim.x + threadIdx.x;
  if (i >= n) return;
  const float v = x[i];
  h16[i] = (_Float16)v;
  hcur[i] = v;
}

__global__ void cast_f16(const float* __restrict__ s, _Float16* __restrict__ d,
                         long long n) {
  const long long i = (long long)blockIdx.x * blockDim.x + threadIdx.x;
  if (i < n) d[i] = (_Float16)s[i];
}

// sinusoid table pe[2L,128]: columns 0..63 sin, 64..127 cos
__global__ void build_pe(float* __restrict__ pe32, _Float16* __restrict__ pe16,
                         int L) {
  const int idx = blockIdx.x * blockDim.x + threadIdx.x;
  const int total = 2 * L * HEAD_DIM;
  if (idx >= total) return;
  const int p = idx >> 7, d = idx & 127;
  const float pos = (float)(p - L);
  const int j = d & 63;
  const float invf = __expf(-(float)j * (9.210340371976184f / 63.f));
  const float ang = pos * invf;
  const float v = (d < 64) ? __sinf(ang) : __cosf(ang);
  pe32[idx] = v;
  pe16[idx] = (_Float16)v;
}

// D_[n,c] = r_w_bias[n,:] · pe[c,:]
__global__ void build_dn(const float* __restrict__ rw,
                         const float* __restrict__ pe32,
                         float* __restrict__ Dn, int L) {
  const int idx = blockIdx.x * blockDim.x + threadIdx.x;
  if (idx >= N_HEAD * 2 * L) return;
  const int nh = idx / (2 * L), c = idx % (2 * L);
  float s = 0.f;
#pragma unroll 8
  for (int d = 0; d < HEAD_DIM; ++d)
    s += rw[nh * HEAD_DIM + d] * pe32[(long long)c * HEAD_DIM + d];
  Dn[idx] = s;
}

// split qv[BL,2048] -> q16[B,H,L,128], qpr16 (= q + r_r_bias), vt16[B,H,128,L]
__global__ void split_qv(const float* __restrict__ qv,
                         const float* __restrict__ rr,   // [H,128] layer slice
                         _Float16* __restrict__ q16,
                         _Float16* __restrict__ qpr16,
                         _Float16* __restrict__ vt16) {
  const long long idx = (long long)blockIdx.x * blockDim.x + threadIdx.x;
  if (idx >= (long long)BL * D_MODEL) return;
  const int d  = (int)(idx % D_MODEL);
  const long long bl = idx / D_MODEL;           // b*L + l
  const int l  = (int)(bl % LL);
  const int b  = (int)(bl / LL);
  const int nh = d >> 7, dd = d & 127;

  const float qe = qv[bl * 2 * D_MODEL + d];
  const float ve = qv[bl * 2 * D_MODEL + D_MODEL + d];

  const long long zq = (((long long)b * N_HEAD + nh) * LL + l) * HEAD_DIM + dd;
  q16[zq]  = (_Float16)qe;
  qpr16[zq] = (_Float16)(qe + rr[nh * HEAD_DIM + dd]);
  vt16[(((long long)b * N_HEAD + nh) * HEAD_DIM + dd) * LL + l] = (_Float16)ve;
}

// ---------------------------------------------------------------------------
// Host launcher
// ---------------------------------------------------------------------------
extern "C" void kernel_launch(void* const* d_in, const int* in_sizes, int n_in,
                              void* d_out, int out_size, void* d_ws, size_t ws_size,
                              hipStream_t stream) {
  (void)in_sizes; (void)n_in; (void)out_size; (void)ws_size;
  const float* x     = (const float*)d_in[0];
  const int*   mask  = (const int*)d_in[1];
  const float* qvw   = (const float*)d_in[2];
  const float* rr    = (const float*)d_in[3];
  const float* rw    = (const float*)d_in[4];
  const float* ln1g  = (const float*)d_in[5];
  const float* ln1b  = (const float*)d_in[6];
  const float* ffnw  = (const float*)d_in[7];
  const float* ffnb  = (const float*)d_in[8];
  const float* ln2g  = (const float*)d_in[9];
  const float* ln2b  = (const float*)d_in[10];
  float* out = (float*)d_out;

  // ---- workspace carve (256B aligned) ----
  char* w = (char*)d_ws;
  auto carve = [&](size_t bytes) -> void* {
    void* p = (void*)w;
    w += (bytes + 255) & ~(size_t)255;
    return p;
  };
  const long long NE   = (long long)BL * D_MODEL;          // 4096*1024
  const long long NQV  = (long long)BL * 2 * D_MODEL;      // 4096*2048
  const long long NHD  = (long long)BB * N_HEAD * LL * HEAD_DIM;
  const long long NSC  = (long long)BB * N_HEAD * LL * LL;

  _Float16* h16    = (_Float16*)carve(NE * 2);
  float*    hcur   = (float*)   carve(NE * 4);
  _Float16* wqv16  = (_Float16*)carve((long long)NUM_LAYERS * 2 * D_MODEL * D_MODEL * 2);
  _Float16* wffn16 = (_Float16*)carve((long long)NUM_LAYERS * D_MODEL * D_MODEL * 2);
  float*    qv32   = (float*)   carve(NQV * 4);
  _Float16* q16    = (_Float16*)carve(NHD * 2);
  _Float16* qpr16  = (_Float16*)carve(NHD * 2);
  _Float16* vt16   = (_Float16*)carve(NHD * 2);
  float*    pe32   = (float*)   carve((long long)L2X * HEAD_DIM * 4);
  _Float16* pe16   = (_Float16*)carve((long long)L2X * HEAD_DIM * 2);
  float*    Dn     = (float*)   carve((long long)N_HEAD * L2X * 4);
  float*    scores = (float*)   carve(NSC * 4);
  _Float16* attn16 = (_Float16*)carve(NSC * 2);
  float*    ctx32  = (float*)   carve(NE * 4);
  float*    h1_32  = (float*)   carve(NE * 4);
  _Float16* h1_16  = (_Float16*)carve(NE * 2);
  float*    ffn32  = (float*)   carve(NE * 4);

  const dim3 blk(256);

  // ---- one-time prep ----
  init_h<<<dim3((unsigned)((NE + 255) / 256)), blk, 0, stream>>>(x, h16, hcur, NE);
  cast_f16<<<dim3((unsigned)((NUM_LAYERS * 2LL * D_MODEL * D_MODEL + 255) / 256)),
             blk, 0, stream>>>(qvw, wqv16, NUM_LAYERS * 2LL * D_MODEL * D_MODEL);
  cast_f16<<<dim3((unsigned)((NUM_LAYERS * (long long)D_MODEL * D_MODEL + 255) / 256)),
             blk, 0, stream>>>(ffnw, wffn16, NUM_LAYERS * (long long)D_MODEL * D_MODEL);
  build_pe<<<dim3((L2X * HEAD_DIM + 255) / 256), blk, 0, stream>>>(pe32, pe16, LL);

  const long long sL128 = (long long)LL * HEAD_DIM;        // 65536
  const long long sHL128 = (long long)N_HEAD * sL128;      // 524288
  const long long sLL = (long long)LL * LL;                // 262144
  const long long sHLL = (long long)N_HEAD * sLL;          // 2097152

  for (int layer = 0; layer < NUM_LAYERS; ++layer) {
    const _Float16* wq = wqv16 + (long long)layer * 2 * D_MODEL * D_MODEL;
    const _Float16* wf = wffn16 + (long long)layer * D_MODEL * D_MODEL;

    // 1) qv = h @ qv_w^T      [4096,2048] = [4096,1024] x [2048,1024]^T
    gemm_nt_wmma<<<dim3(2 * D_MODEL / 64, BL / 256, 1), blk, 0, stream>>>(
        h16, D_MODEL, 0, 0, wq, D_MODEL, 0, 0, qv32, 2 * D_MODEL, 0, 0,
        BL, 2 * D_MODEL, D_MODEL, 1, nullptr, 0);

    // 2) split/pack: q16, q+rr, V^T (all f16)
    split_qv<<<dim3((unsigned)(NE / 256)), blk, 0, stream>>>(
        qv32, rr + (long long)layer * N_HEAD * HEAD_DIM, q16, qpr16, vt16);

    // 3) D_[n,c] = rw · pe
    build_dn<<<dim3((N_HEAD * L2X + 255) / 256), blk, 0, stream>>>(
        rw + (long long)layer * N_HEAD * HEAD_DIM, pe32, Dn, LL);

    // 4) AC scores = (q+rr) @ K^T, K is raw h per head (row stride 1024)
    gemm_nt_wmma<<<dim3(LL / 64, LL / 256, BB * N_HEAD), blk, 0, stream>>>(
        qpr16, HEAD_DIM, sHL128, sL128,
        h16, D_MODEL, (long long)LL * D_MODEL, HEAD_DIM,
        scores, LL, sHLL, sLL,
        LL, LL, HEAD_DIM, N_HEAD, nullptr, 0);

    // 5) B_ GEMM + folded shift: scores[i,c-L+i] += q·pe[c] + D_[n,c]
    relb_shift_add<<<dim3(L2X / 64, LL / 256, BB * N_HEAD), blk, 0, stream>>>(
        q16, pe16, Dn, scores, LL, N_HEAD);

    // 6) masked softmax -> f16 probabilities
    softmax_row<<<dim3(BB * N_HEAD * LL), blk, 0, stream>>>(
        scores, mask, attn16, LL, N_HEAD);

    // 7) ctx = attn @ V  (NT against V^T[128,L]); write [B,L,H*128]
    gemm_nt_wmma<<<dim3(HEAD_DIM / 64, LL / 256, BB * N_HEAD), blk, 0, stream>>>(
        attn16, LL, sHLL, sLL,
        vt16, LL, sHL128, sL128,
        ctx32, D_MODEL, (long long)LL * D_MODEL, HEAD_DIM,
        LL, HEAD_DIM, LL, N_HEAD, nullptr, 0);

    // 8) h1 = LN(ctx + h)
    add_layernorm<<<dim3(BL), blk, 0, stream>>>(
        ctx32, hcur, ln1g + (long long)layer * D_MODEL,
        ln1b + (long long)layer * D_MODEL, h1_32, h1_16, D_MODEL);

    // 9) ffn = relu(h1 @ ffn_w^T + b)
    gemm_nt_wmma<<<dim3(D_MODEL / 64, BL / 256, 1), blk, 0, stream>>>(
        h1_16, D_MODEL, 0, 0, wf, D_MODEL, 0, 0, ffn32, D_MODEL, 0, 0,
        BL, D_MODEL, D_MODEL, 1, ffnb + (long long)layer * D_MODEL, 1);

    // 10) h = LN(h1 + ffn); last layer writes d_out, also refresh h16
    float* o32 = (layer == NUM_LAYERS - 1) ? out : hcur;
    add_layernorm<<<dim3(BL), blk, 0, stream>>>(
        h1_32, ffn32, ln2g + (long long)layer * D_MODEL,
        ln2b + (long long)layer * D_MODEL, o32, h16, D_MODEL);
  }
}
